// RelaComMultiheadAttention_83451214561343
// MI455X (gfx1250) — matrix operand order repaired and checked by reading, not compiled, not run on hardware
//
#include <hip/hip_runtime.h>
#include <hip/hip_bf16.h>

// ---------------------------------------------------------------------------
// RelaComMultiheadAttention on MI455X (gfx1250, wave32, WMMA)
//
// Shapes: H=8 heads, D=64, N=2, Q=P=128, B=4.
// Key optimization: the per-head projections of the huge emb tensor
// (emb @ W^T then dotted with a per-(n,q,b) query vector) are folded into the
// query side:  <emb @ W^T, v> = <emb, W^T v>.  BD then becomes, per (n,q,b),
// a [P=128, K=128] x [K=128, 16] f16 WMMA GEMM (16 cols = 8 heads x {r,i}),
// done once with emb[n,q,:,b,:] (straight) and once with emb[n,:,q,b,:]
// (swapped).  emb is read exactly twice -> memory-bound at ~128MB of traffic.
// AC scores and the attention*V product are also v_wmma_f32_16x16x32_f16.
// ---------------------------------------------------------------------------

#define HH 8
#define DD 64
#define NN 2
#define QQ 128
#define PP 128
#define BB 4
#define RR (QQ*BB)          // 512 rows per (h,n)
#define TEMP_F 30.0f
#define EPS_F 1e-5f

typedef __attribute__((ext_vector_type(16))) _Float16 v16h;
typedef __attribute__((ext_vector_type(8)))  float    v8f;

__device__ __forceinline__ v8f wmma_f16(v16h a, v16h b, v8f c) {
  // D = A(16x32 f16) * B(32x16 f16) + C(16x16 f32)
  return __builtin_amdgcn_wmma_f32_16x16x32_f16(
      /*neg_a=*/false, a, /*neg_b=*/false, b,
      /*c_mod=*/(short)0, c, /*reuse_a=*/false, /*reuse_b=*/false);
}

// ===========================================================================
// K1: complex projections of Q and K + instance norm.
// grid.x = H*N*2 (which: 0=query, 1=key), 256 threads (thread = (rowgrp, d)).
// Writes f16 ww-normalized queries, f32 wr-normalized queries, f16 norm keys.
// ===========================================================================
__global__ void __launch_bounds__(256) k1_projnorm(
    const float* __restrict__ qr, const float* __restrict__ qi,
    const float* __restrict__ kr, const float* __restrict__ ki,
    const float* __restrict__ WQr, const float* __restrict__ WQrb,
    const float* __restrict__ WQi, const float* __restrict__ WQib,
    const float* __restrict__ WKr, const float* __restrict__ WKrb,
    const float* __restrict__ WKi, const float* __restrict__ WKib,
    const float* __restrict__ wwrg, const float* __restrict__ wwrb,
    const float* __restrict__ wwig, const float* __restrict__ wwib,
    const float* __restrict__ wrrg, const float* __restrict__ wrrb,
    const float* __restrict__ wrig, const float* __restrict__ wrib,
    float* __restrict__ preBuf,
    _Float16* __restrict__ wwqr16, _Float16* __restrict__ wwqi16,
    float* __restrict__ wrqr32, float* __restrict__ wrqi32,
    _Float16* __restrict__ wkr16, _Float16* __restrict__ wki16)
{
  const int bx    = blockIdx.x;     // 0..31
  const int which = bx & 1;         // 0=query 1=key
  const int hn    = bx >> 1;        // h*N + n
  const int h     = hn >> 1;
  const int n     = hn & 1;
  const int t     = threadIdx.x;
  const int d     = t & 63;
  const int rg    = t >> 6;         // 0..3 row groups of 128 rows

  __shared__ float sWrT[64 * 64];   // W transposed: [e][d] for conflict-free d
  __shared__ float sWiT[64 * 64];
  __shared__ float sRed[4][64][4];
  __shared__ float sMR[64], sIR[64], sMI[64], sII[64];

  const float* Wr = (which == 0 ? WQr  : WKr)  + h * 4096;
  const float* Wi = (which == 0 ? WQi  : WKi)  + h * 4096;
  const float* br = (which == 0 ? WQrb : WKrb) + h * 64;
  const float* bi = (which == 0 ? WQib : WKib) + h * 64;
  const float* Xr = (which == 0 ? qr : kr) + (size_t)n * RR * DD;
  const float* Xi = (which == 0 ? qi : ki) + (size_t)n * RR * DD;

  for (int idx = t; idx < 4096; idx += 256) {
    int dout = idx >> 6, ein = idx & 63;
    sWrT[ein * 64 + dout] = Wr[idx];
    sWiT[ein * 64 + dout] = Wi[idx];
  }
  __syncthreads();

  float* preR = preBuf + (size_t)bx * 2 * RR * DD;
  float* preI = preR + RR * DD;
  const float br_d = br[d] - bi[d];   // bias of (p11 - p22)
  const float bi_d = br[d] + bi[d];   // bias of (p12 + p21)

  float s1r = 0.f, s2r = 0.f, s1i = 0.f, s2i = 0.f;
  for (int j = 0; j < 128; ++j) {
    const int r = rg * 128 + j;
    const float* xr = Xr + (size_t)r * DD;   // same addr across lanes of a wave
    const float* xi = Xi + (size_t)r * DD;   // -> hardware broadcast
    float ar = br_d, ai = bi_d;
    for (int e = 0; e < 64; ++e) {
      const float vr = xr[e], vi = xi[e];
      const float wr_ = sWrT[e * 64 + d], wi_ = sWiT[e * 64 + d];
      ar += vr * wr_ - vi * wi_;
      ai += vr * wi_ + vi * wr_;
    }
    preR[(size_t)r * DD + d] = ar;
    preI[(size_t)r * DD + d] = ai;
    s1r += ar; s2r += ar * ar; s1i += ai; s2i += ai * ai;
  }
  sRed[rg][d][0] = s1r; sRed[rg][d][1] = s2r;
  sRed[rg][d][2] = s1i; sRed[rg][d][3] = s2i;
  __syncthreads();
  if (rg == 0) {
    float a0 = 0.f, a1 = 0.f, a2 = 0.f, a3 = 0.f;
    for (int g = 0; g < 4; ++g) {
      a0 += sRed[g][d][0]; a1 += sRed[g][d][1];
      a2 += sRed[g][d][2]; a3 += sRed[g][d][3];
    }
    const float mr = a0 * (1.f / RR), mi = a2 * (1.f / RR);
    const float vr = a1 * (1.f / RR) - mr * mr;
    const float vi = a3 * (1.f / RR) - mi * mi;
    sMR[d] = mr; sIR[d] = rsqrtf(vr + EPS_F);
    sMI[d] = mi; sII[d] = rsqrtf(vi + EPS_F);
  }
  __syncthreads();
  const float mr = sMR[d], ivr = sIR[d], mi = sMI[d], ivi = sII[d];

  float gwr = 0.f, bwr = 0.f, gwi = 0.f, bwi = 0.f;
  float grr = 0.f, brr = 0.f, gri = 0.f, bri = 0.f;
  if (which == 0) {
    gwr = wwrg[h * 64 + d]; bwr = wwrb[h * 64 + d];
    gwi = wwig[h * 64 + d]; bwi = wwib[h * 64 + d];
    grr = wrrg[h * 64 + d]; brr = wrrb[h * 64 + d];
    gri = wrig[h * 64 + d]; bri = wrib[h * 64 + d];
  }
  const size_t ob = (size_t)hn * RR * DD;
  for (int j = 0; j < 128; ++j) {
    const int r = rg * 128 + j;
    const float nr = (preR[(size_t)r * DD + d] - mr) * ivr;
    const float ni = (preI[(size_t)r * DD + d] - mi) * ivi;
    const size_t o = ob + (size_t)r * DD + d;
    if (which == 0) {
      wwqr16[o] = (_Float16)(nr * gwr + bwr);
      wwqi16[o] = (_Float16)(ni * gwi + bwi);
      wrqr32[o] = nr * grr + brr;
      wrqi32[o] = ni * gri + bri;
    } else {
      wkr16[o] = (_Float16)nr;   // instnorm affine=False for keys
      wki16[o] = (_Float16)ni;
    }
  }
}

// ===========================================================================
// K1b: A-vectors  A1=WKRr^T wr_qr, A2=WKRi^T wr_qr, A3=WKRr^T wr_qi,
// A4=WKRi^T wr_qi  (per h,n,q,b) + bias scalars s_r, s_i.
// grid.x = H*N, 256 threads.
// ===========================================================================
__global__ void __launch_bounds__(256) k1b_avec(
    const float* __restrict__ WKRr, const float* __restrict__ WKRrb,
    const float* __restrict__ WKRi, const float* __restrict__ WKRib,
    const float* __restrict__ wrqr32, const float* __restrict__ wrqi32,
    _Float16* __restrict__ A1, _Float16* __restrict__ A2,
    _Float16* __restrict__ A3, _Float16* __restrict__ A4,
    float* __restrict__ sbr, float* __restrict__ sbi)
{
  const int hn = blockIdx.x;
  const int h  = hn >> 1;
  const int t  = threadIdx.x;
  const int d  = t & 63;
  const int rg = t >> 6;

  __shared__ float sWr[4096], sWi[4096], sbd[64], sbs[64];
  for (int idx = t; idx < 4096; idx += 256) {
    sWr[idx] = WKRr[h * 4096 + idx];   // [e][d] row-major: d consecutive
    sWi[idx] = WKRi[h * 4096 + idx];
  }
  if (t < 64) {
    const float a = WKRrb[h * 64 + t], b = WKRib[h * 64 + t];
    sbd[t] = a - b; sbs[t] = a + b;
  }
  __syncthreads();

  const size_t base = (size_t)hn * RR * DD;
  for (int j = 0; j < 128; ++j) {
    const int r = rg * 128 + j;
    const float* xr = wrqr32 + base + (size_t)r * DD;  // broadcast reads
    const float* xi = wrqi32 + base + (size_t)r * DD;
    float a1 = 0.f, a2 = 0.f, a3 = 0.f, a4 = 0.f, s_r = 0.f, s_i = 0.f;
    for (int e = 0; e < 64; ++e) {
      const float vr = xr[e], vi = xi[e];
      const float wr_ = sWr[e * 64 + d], wi_ = sWi[e * 64 + d];
      a1 += wr_ * vr; a2 += wi_ * vr; a3 += wr_ * vi; a4 += wi_ * vi;
      s_r += sbd[e] * vr - sbs[e] * vi;
      s_i += sbd[e] * vi + sbs[e] * vr;
    }
    const size_t o = base + (size_t)r * DD + d;
    A1[o] = (_Float16)a1; A2[o] = (_Float16)a2;
    A3[o] = (_Float16)a3; A4[o] = (_Float16)a4;
    if (d == 0) { sbr[(size_t)hn * RR + r] = s_r; sbi[(size_t)hn * RR + r] = s_i; }
  }
}

// ===========================================================================
// K2: BD scores via WMMA while streaming emb (THE memory-bound stage).
// grid.x = N*Q*B = 1024 blocks; 128 threads = 4 waves; wave handles 2 p-tiles.
// Per (n,q,b):  BD[p, 16cols] += E_part[p,128] (f32->f16) @ Coef[128,16] (f16)
// over part in {straight, swapped}; then + per-column bias scalar.
// BD_r = Er*A1 - Ei*A2  (straight)  - Er'*A4 - Ei'*A3  (swapped)  + s_r
// BD_i = Er*A3 - Ei*A4  (straight)  + Er'*A2 + Ei'*A1  (swapped)  + s_i
// ===========================================================================
__global__ void __launch_bounds__(128) k2_bd(
    const float* __restrict__ Er, const float* __restrict__ Ei,
    const _Float16* __restrict__ A1, const _Float16* __restrict__ A2,
    const _Float16* __restrict__ A3, const _Float16* __restrict__ A4,
    const float* __restrict__ sbr, const float* __restrict__ sbi,
    float* __restrict__ BD)
{
  const int bx   = blockIdx.x;          // n*512 + q*4 + b
  const int n    = bx >> 9;
  const int rem  = bx & 511;            // q*B + b
  const int q    = rem >> 2;
  const int b    = rem & 3;
  const int lane = threadIdx.x & 31;
  const int wave = threadIdx.x >> 5;
  const int mrow = lane & 15;
  const int kh   = lane >> 4;           // 0 or 1 (lane half)
  const int col  = lane & 15;           // output column: 2*head + ri
  const int ch   = col >> 1;            // head for this column
  const int cri  = col & 1;             // 0 = BD_r, 1 = BD_i

  // Coefficient-vector row base for this (head, n, q, b)
  const size_t vbase = ((size_t)(ch * NN + n) * RR + rem) * DD;

  for (int pt = wave * 2; pt < wave * 2 + 2; ++pt) {
    v8f acc = {};
    const int prow = pt * 16 + mrow;
    const size_t rowS = ((((size_t)n * QQ + q)    * PP + prow) * BB + b) * DD; // straight
    const size_t rowW = ((((size_t)n * QQ + prow) * PP + q)    * BB + b) * DD; // swapped
    __builtin_prefetch(Er + rowW, 0, 1);   // global_prefetch_b8
    __builtin_prefetch(Ei + rowW, 0, 1);

#pragma unroll
    for (int part = 0; part < 2; ++part) {
      const size_t rowbase = part == 0 ? rowS : rowW;
#pragma unroll
      for (int ks = 0; ks < 4; ++ks) {
        // ---- A fragment: emb row, f32 -> f16 on the fly ----
        v16h a;
#pragma unroll
        for (int c2 = 0; c2 < 2; ++c2) {
          const int kk = ks * 32 + kh * 16 + c2 * 8;      // K base of 8-chunk
          const float* src = (kk < 64 ? Er : Ei) + rowbase + (kk & 63);
#pragma unroll
          for (int i = 0; i < 8; ++i) a[c2 * 8 + i] = (_Float16)src[i];
        }
        // ---- B fragment: coefficient vector (sign/select per column) ----
        const int kb  = ks * 32 + kh * 16;   // 16-aligned, never straddles 64
        const int khi = kb >> 6;             // uniform per ks
        const int sel = part == 0 ? (cri == 0 ? (khi ? 1 : 0) : (khi ? 3 : 2))
                                  : (cri == 0 ? (khi ? 2 : 3) : (khi ? 0 : 1));
        const bool neg = (part == 0) ? (khi != 0) : (cri == 0);
        const _Float16* vp =
            (sel == 0 ? A1 : sel == 1 ? A2 : sel == 2 ? A3 : A4) + vbase + (kb & 63);
        v16h bf;
#pragma unroll
        for (int i = 0; i < 16; ++i) {
          const _Float16 v = vp[i];
          bf[i] = neg ? (_Float16)(-v) : v;
        }
        acc = wmma_f16(a, bf, acc);
      }
    }
    const float s = (cri ? sbi : sbr)[(size_t)(ch * NN + n) * RR + rem];
#pragma unroll
    for (int j = 0; j < 8; ++j) {
      const int pr = pt * 16 + kh * 8 + j;
      BD[((size_t)bx * PP + pr) * 16 + col] = acc[j] + s;
    }
  }
}

// ===========================================================================
// K3: AC scores via WMMA.  grid = (H*N*B, 8 q-tiles), 256 thr = 8 waves
// (wave = p-tile).  A row q = [wwqr | wwqi], Br col p = [wkr | -wki],
// Bi col p = [wki | wkr], K=128.
// ===========================================================================
__global__ void __launch_bounds__(256) k3_ac(
    const _Float16* __restrict__ wwqr16, const _Float16* __restrict__ wwqi16,
    const _Float16* __restrict__ wkr16, const _Float16* __restrict__ wki16,
    float* __restrict__ ACr, float* __restrict__ ACi)
{
  const int c  = blockIdx.x;            // h*8 + n*4 + b  == (h*N+n)*B + b
  const int h  = c >> 3;
  const int n  = (c >> 2) & 1;
  const int b  = c & 3;
  const int mt = blockIdx.y;
  const int lane = threadIdx.x & 31;
  const int pt   = threadIdx.x >> 5;    // p-tile
  const int mrow = lane & 15;
  const int kh   = lane >> 4;

  const size_t hb = (size_t)(h * NN + n) * RR * DD;  // [hn][r][64]
  const int qrow = mt * 16 + mrow;
  const int prow = pt * 16 + (lane & 15);
  const size_t qoff = hb + ((size_t)qrow * BB + b) * DD;
  const size_t poff = hb + ((size_t)prow * BB + b) * DD;

  v8f accR = {}, accI = {};
#pragma unroll
  for (int ks = 0; ks < 4; ++ks) {
    v16h a;
#pragma unroll
    for (int c2 = 0; c2 < 2; ++c2) {
      const int kk = ks * 32 + kh * 16 + c2 * 8;
      const _Float16* src = (kk < 64 ? wwqr16 : wwqi16) + qoff + (kk & 63);
#pragma unroll
      for (int i = 0; i < 8; ++i) a[c2 * 8 + i] = src[i];
    }
    const int kb  = ks * 32 + kh * 16;
    const int khi = kb >> 6;
    const _Float16* pr_ = (khi ? wki16 : wkr16) + poff + (kb & 63);
    const _Float16* pi_ = (khi ? wkr16 : wki16) + poff + (kb & 63);
    const bool negR = (khi != 0);       // AC_r: high-K uses -wk_i
    v16h bR, bI;
#pragma unroll
    for (int i = 0; i < 16; ++i) {
      const _Float16 vr = pr_[i];
      bR[i] = negR ? (_Float16)(-vr) : vr;
      bI[i] = pi_[i];
    }
    accR = wmma_f16(a, bR, accR);
    accI = wmma_f16(a, bI, accI);
  }
#pragma unroll
  for (int j = 0; j < 8; ++j) {
    const int qr_ = mt * 16 + kh * 8 + j;
    const size_t o = ((size_t)c * QQ + qr_) * PP + prow;  // [h][n][b][q][p]
    ACr[o] = accR[j];
    ACi[o] = accI[j];
  }
}

// ===========================================================================
// K4: aff = sqrt(ACr^2+ACi^2)+sqrt(BDr^2+BDi^2); softmax over b (axis=-1!).
// One thread per (h,n,q,p); handles all 4 b.  Writes f16 probabilities.
// ===========================================================================
__global__ void __launch_bounds__(256) k4_aff(
    const float* __restrict__ ACr, const float* __restrict__ ACi,
    const float* __restrict__ BD, _Float16* __restrict__ aff16)
{
  const int idx = blockIdx.x * 256 + threadIdx.x;   // H*N*Q*P = 262144
  const int p = idx & 127;
  const int q = (idx >> 7) & 127;
  const int n = (idx >> 14) & 1;
  const int h = idx >> 15;
  const int hn = h * NN + n;

  float e4[4];
  float m = -1e30f;
#pragma unroll
  for (int b = 0; b < 4; ++b) {
    const size_t oA = (((size_t)hn * BB + b) * QQ + q) * PP + p;
    const float acr = ACr[oA], aci = ACi[oA];
    const size_t oB = (((size_t)n * RR + q * BB + b) * PP + p) * 16;
    const float bdr = BD[oB + h * 2], bdi = BD[oB + h * 2 + 1];
    float a = sqrtf(acr * acr + aci * aci) + sqrtf(bdr * bdr + bdi * bdi);
    a *= TEMP_F;
    e4[b] = a;
    m = fmaxf(m, a);
  }
  float s = 0.f;
#pragma unroll
  for (int b = 0; b < 4; ++b) { e4[b] = __expf(e4[b] - m); s += e4[b]; }
  const float inv = 1.f / s;
#pragma unroll
  for (int b = 0; b < 4; ++b)
    aff16[(((size_t)hn * BB + b) * QQ + q) * PP + p] = (_Float16)(e4[b] * inv);
}

// ===========================================================================
// K4b: V -> f16, transposed to [n][b][d][p] so K5's B-side reads 16
// contiguous K(=p) values per column d.
// ===========================================================================
__global__ void __launch_bounds__(256) k4b_vconv(
    const float* __restrict__ Vr, const float* __restrict__ Vi,
    _Float16* __restrict__ V16r, _Float16* __restrict__ V16i)
{
  const int idx = blockIdx.x * 256 + threadIdx.x;  // N*B*D*P = 65536
  const int p = idx & 127;
  const int d = (idx >> 7) & 63;
  const int b = (idx >> 13) & 3;
  const int n = idx >> 15;
  const size_t src = (((size_t)n * PP + p) * BB + b) * DD + d;
  V16r[idx] = (_Float16)Vr[src];
  V16i[idx] = (_Float16)Vi[src];
}

// ===========================================================================
// K5: out[n,q,b,h*64+d] = sum_p aff[h,n,q,p,b] * V[n,p,b,d]  via WMMA.
// grid = (H*N*B, 8 q-tiles), 128 thr = 4 waves (wave = d-tile).
// ===========================================================================
__global__ void __launch_bounds__(128) k5_av(
    const _Float16* __restrict__ aff16,
    const _Float16* __restrict__ V16r, const _Float16* __restrict__ V16i,
    float* __restrict__ outR, float* __restrict__ outI)
{
  const int c  = blockIdx.x;            // h*8 + n*4 + b
  const int h  = c >> 3;
  const int n  = (c >> 2) & 1;
  const int b  = c & 3;
  const int mt = blockIdx.y;
  const int lane = threadIdx.x & 31;
  const int dt   = threadIdx.x >> 5;    // d-tile 0..3
  const int mrow = lane & 15;
  const int kh   = lane >> 4;

  const int qrow = mt * 16 + mrow;
  const int dcol = dt * 16 + (lane & 15);
  const size_t arow = ((size_t)c * QQ + qrow) * PP;
  const size_t brow = (((size_t)n * BB + b) * DD + dcol) * PP;

  v8f accR = {}, accI = {};
#pragma unroll
  for (int ks = 0; ks < 4; ++ks) {
    v16h a;
#pragma unroll
    for (int c2 = 0; c2 < 2; ++c2) {
      const int p0 = ks * 32 + kh * 16 + c2 * 8;
      const _Float16* src = aff16 + arow + p0;
#pragma unroll
      for (int i = 0; i < 8; ++i) a[c2 * 8 + i] = src[i];
    }
    const int pb = ks * 32 + kh * 16;
    const _Float16* srcR = V16r + brow + pb;
    const _Float16* srcI = V16i + brow + pb;
    v16h bR, bI;
#pragma unroll
    for (int i = 0; i < 16; ++i) { bR[i] = srcR[i]; bI[i] = srcI[i]; }
    accR = wmma_f16(a, bR, accR);
    accI = wmma_f16(a, bI, accI);
  }
#pragma unroll
  for (int j = 0; j < 8; ++j) {
    const int qr_ = mt * 16 + kh * 8 + j;
    const size_t o = (((size_t)n * QQ + qr_) * BB + b) * (HH * DD) + h * DD + dcol;
    outR[o] = accR[j];
    outI[o] = accI[j];
  }
}

// ===========================================================================
extern "C" void kernel_launch(void* const* d_in, const int* in_sizes, int n_in,
                              void* d_out, int out_size, void* d_ws, size_t ws_size,
                              hipStream_t stream) {
  const float* qr  = (const float*)d_in[0];
  const float* qi  = (const float*)d_in[1];
  const float* kr  = (const float*)d_in[2];
  const float* ki  = (const float*)d_in[3];
  const float* vr  = (const float*)d_in[4];
  const float* vi  = (const float*)d_in[5];
  const float* er  = (const float*)d_in[6];
  const float* ei  = (const float*)d_in[7];
  const float* WKr_w  = (const float*)d_in[8];
  const float* WKr_b  = (const float*)d_in[9];
  const float* WKi_w  = (const float*)d_in[10];
  const float* WKi_b  = (const float*)d_in[11];
  const float* WKRr_w = (const float*)d_in[12];
  const float* WKRr_b = (const float*)d_in[13];
  const float* WKRi_w = (const float*)d_in[14];
  const float* WKRi_b = (const float*)d_in[15];
  const float* WQr_w  = (const float*)d_in[16];
  const float* WQr_b  = (const float*)d_in[17];
  const float* WQi_w  = (const float*)d_in[18];
  const float* WQi_b  = (const float*)d_in[19];
  const float* ww_r_g  = (const float*)d_in[20];
  const float* ww_r_bt = (const float*)d_in[21];
  const float* ww_i_g  = (const float*)d_in[22];
  const float* ww_i_bt = (const float*)d_in[23];
  const float* wr_r_g  = (const float*)d_in[24];
  const float* wr_r_bt = (const float*)d_in[25];
  const float* wr_i_g  = (const float*)d_in[26];
  const float* wr_i_bt = (const float*)d_in[27];

  // ---- workspace carve-out (~38.5 MB) ----
  char* ws = (char*)d_ws;
  size_t off = 0;
  auto carve = [&](size_t bytes) -> void* {
    void* p = ws + off;
    off += (bytes + 255) & ~(size_t)255;
    return p;
  };
  float*    preBuf  = (float*)   carve((size_t)32 * 2 * RR * DD * 4);  // 8 MB
  float*    wrqr32  = (float*)   carve((size_t)16 * RR * DD * 4);      // 2 MB
  float*    wrqi32  = (float*)   carve((size_t)16 * RR * DD * 4);
  _Float16* wwqr16  = (_Float16*)carve((size_t)16 * RR * DD * 2);      // 1 MB
  _Float16* wwqi16  = (_Float16*)carve((size_t)16 * RR * DD * 2);
  _Float16* wkr16   = (_Float16*)carve((size_t)16 * RR * DD * 2);
  _Float16* wki16   = (_Float16*)carve((size_t)16 * RR * DD * 2);
  _Float16* A1      = (_Float16*)carve((size_t)16 * RR * DD * 2);
  _Float16* A2      = (_Float16*)carve((size_t)16 * RR * DD * 2);
  _Float16* A3      = (_Float16*)carve((size_t)16 * RR * DD * 2);
  _Float16* A4      = (_Float16*)carve((size_t)16 * RR * DD * 2);
  float*    sbr     = (float*)   carve((size_t)16 * RR * 4);
  float*    sbi     = (float*)   carve((size_t)16 * RR * 4);
  float*    ACr     = (float*)   carve((size_t)64 * QQ * PP * 4);      // 4 MB
  float*    ACi     = (float*)   carve((size_t)64 * QQ * PP * 4);
  float*    BD      = (float*)   carve((size_t)NN * RR * PP * 16 * 4); // 8 MB
  _Float16* aff16   = (_Float16*)carve((size_t)64 * QQ * PP * 2);      // 2 MB
  _Float16* V16r    = (_Float16*)carve((size_t)NN * BB * DD * PP * 2);
  _Float16* V16i    = (_Float16*)carve((size_t)NN * BB * DD * PP * 2);
  (void)ws_size; (void)in_sizes; (void)n_in; (void)out_size;

  float* outR = (float*)d_out;
  float* outI = outR + (size_t)NN * QQ * BB * HH * DD;   // 524288

  k1_projnorm<<<32, 256, 0, stream>>>(
      qr, qi, kr, ki,
      WQr_w, WQr_b, WQi_w, WQi_b, WKr_w, WKr_b, WKi_w, WKi_b,
      ww_r_g, ww_r_bt, ww_i_g, ww_i_bt, wr_r_g, wr_r_bt, wr_i_g, wr_i_bt,
      preBuf, wwqr16, wwqi16, wrqr32, wrqi32, wkr16, wki16);

  k1b_avec<<<16, 256, 0, stream>>>(
      WKRr_w, WKRr_b, WKRi_w, WKRi_b, wrqr32, wrqi32,
      A1, A2, A3, A4, sbr, sbi);

  k2_bd<<<NN * QQ * BB, 128, 0, stream>>>(
      er, ei, A1, A2, A3, A4, sbr, sbi, BD);

  k3_ac<<<dim3(64, 8), 256, 0, stream>>>(
      wwqr16, wwqi16, wkr16, wki16, ACr, ACi);

  k4b_vconv<<<256, 256, 0, stream>>>(vr, vi, V16r, V16i);

  k4_aff<<<1024, 256, 0, stream>>>(ACr, ACi, BD, aff16);

  k5_av<<<dim3(64, 8), 128, 0, stream>>>(aff16, V16r, V16i, outR, outI);
}
